// Palm_6605659701455
// MI455X (gfx1250) — matrix operand order repaired and checked by reading, not compiled, not run on hardware
//
#include <hip/hip_runtime.h>
#include <hip/hip_bf16.h>
#include <cstdint>
#include <cstddef>

// ---------------------------------------------------------------------------
// Model constants (from reference)
// ---------------------------------------------------------------------------
#define E_DIM   1024
#define H_HEADS 16
#define HD_DIM  64
#define L_LAYERS 4
#define NH_DIM  4096
#define V_VOCAB 32000
#define T_SEQ   1024
#define ATTN_SCALE 0.12f
#define RMS_EPS 1.19209290e-07f

typedef __attribute__((ext_vector_type(16))) _Float16 v16h;
typedef __attribute__((ext_vector_type(8)))  float    v8f;

union Frag { uint4 q[2]; v16h h; };

// Async global->LDS copy (ASYNCcnt path), guarded so the build stays green on
// toolchains that don't declare the builtin.
#if defined(__HIP_DEVICE_COMPILE__) && \
    __has_builtin(__builtin_amdgcn_global_load_async_to_lds_b128)
#define HAVE_ASYNC_LDS 1
#else
#define HAVE_ASYNC_LDS 0
#endif

#if HAVE_ASYNC_LDS
// Builtin signature (from compiler diagnostics): param0 is a non-const
// int4 (gcc vector_size(16)) pointer in the global (AS1/"__device__") address
// space; param1 is the LDS-side pointer; params 2/3 are imm offset / cpol.
typedef int async_i4 __attribute__((vector_size(16)));
#define ASYNC_G(p) \
    ((__attribute__((address_space(1))) async_i4*)(void*)(uintptr_t)(p))
#define ASYNC_L(p) \
    ((__attribute__((address_space(3))) async_i4*)(void*)(p))
#endif

__device__ __forceinline__ void wait_async_zero() {
#if defined(__HIP_DEVICE_COMPILE__)
#if __has_builtin(__builtin_amdgcn_s_wait_asynccnt)
    __builtin_amdgcn_s_wait_asynccnt(0);
#else
    asm volatile("s_wait_asynccnt 0" ::: "memory");
#endif
#endif
}

// ---------------------------------------------------------------------------
// Block-wide reductions (256 threads)
// ---------------------------------------------------------------------------
__device__ __forceinline__ float block_reduce_sum256(float v, float* sh) {
    int tid = threadIdx.x;
    sh[tid] = v; __syncthreads();
    for (int s = 128; s > 0; s >>= 1) {
        if (tid < s) sh[tid] += sh[tid + s];
        __syncthreads();
    }
    float r = sh[0];
    __syncthreads();
    return r;
}

__device__ __forceinline__ float block_reduce_max256(float v, float* sh) {
    int tid = threadIdx.x;
    sh[tid] = v; __syncthreads();
    for (int s = 128; s > 0; s >>= 1) {
        if (tid < s) sh[tid] = fmaxf(sh[tid], sh[tid + s]);
        __syncthreads();
    }
    float r = sh[0];
    __syncthreads();
    return r;
}

// ---------------------------------------------------------------------------
// Embedding: x[t] = wte[idx[t]] + wpe[t]
// ---------------------------------------------------------------------------
__global__ __launch_bounds__(256) void embed_kernel(
    const int* __restrict__ idx, const float* __restrict__ wte,
    const float* __restrict__ wpe, float* __restrict__ x)
{
    int t = blockIdx.x;
    int token = idx[t];
    const float* we = wte + (size_t)token * E_DIM;
    const float* pe = wpe + (size_t)t * E_DIM;
    float* xo = x + (size_t)t * E_DIM;
    for (int e = threadIdx.x; e < E_DIM; e += 256)
        xo[e] = we[e] + pe[e];
}

// ---------------------------------------------------------------------------
// Row-wise RMS norm (no weight), row length K. out = in * rsqrt(mean(in^2)+eps)
// ---------------------------------------------------------------------------
__global__ __launch_bounds__(256) void rms_rows_kernel(
    const float* __restrict__ in, float* __restrict__ out, int K)
{
    __shared__ float sh[256];
    int row = blockIdx.x;
    const float* p = in + (size_t)row * K;
    float ss = 0.f;
    for (int i = threadIdx.x; i < K; i += 256) { float v = p[i]; ss += v * v; }
    ss = block_reduce_sum256(ss, sh);
    float r = rsqrtf(ss / (float)K + RMS_EPS);
    float* o = out + (size_t)row * K;
    for (int i = threadIdx.x; i < K; i += 256) o[i] = p[i] * r;
}

// ---------------------------------------------------------------------------
// BitLinear activation path, fused: rms -> absmax int8 fake-quant -> f16
// ---------------------------------------------------------------------------
__global__ __launch_bounds__(256) void actquant_kernel(
    const float* __restrict__ in, _Float16* __restrict__ out, int K)
{
    __shared__ float sh[256];
    int row = blockIdx.x;
    const float* p = in + (size_t)row * K;
    float ss = 0.f;
    for (int i = threadIdx.x; i < K; i += 256) { float v = p[i]; ss += v * v; }
    ss = block_reduce_sum256(ss, sh);
    float r = rsqrtf(ss / (float)K + RMS_EPS);
    float mx = 0.f;
    for (int i = threadIdx.x; i < K; i += 256) mx = fmaxf(mx, fabsf(p[i] * r));
    mx = block_reduce_max256(mx, sh);
    float s = 127.0f / fmaxf(mx, 1e-5f);
    float inv_s = 1.0f / s;
    _Float16* o = out + (size_t)row * K;
    for (int i = threadIdx.x; i < K; i += 256) {
        float xn = p[i] * r;
        float q = fminf(fmaxf(roundf(xn * s), -128.f), 127.f) * inv_s;
        o[i] = (_Float16)q;
    }
}

// ---------------------------------------------------------------------------
// Weight fake-quant: mean(|w|) reduce, then ternary quant -> f16
// ---------------------------------------------------------------------------
__global__ __launch_bounds__(256) void zero_kernel(float* p, int n) {
    int i = blockIdx.x * blockDim.x + threadIdx.x;
    if (i < n) p[i] = 0.f;
}

__global__ __launch_bounds__(256) void wabssum_kernel(
    const float* __restrict__ w, long n, float* __restrict__ sum)
{
    __shared__ float sh[256];
    float s = 0.f;
    for (long i = (long)blockIdx.x * 256 + threadIdx.x; i < n;
         i += (long)gridDim.x * 256)
        s += fabsf(w[i]);
    s = block_reduce_sum256(s, sh);
    if (threadIdx.x == 0) atomicAdd(sum, s);
}

__global__ __launch_bounds__(256) void wquant_kernel(
    const float* __restrict__ w, const float* __restrict__ sumabs, long n,
    _Float16* __restrict__ out)
{
    float mean = *sumabs / (float)n;
    float ws = 1.0f / fmaxf(mean, 1e-5f);
    float inv = 1.0f / ws;
    for (long i = (long)blockIdx.x * 256 + threadIdx.x; i < n;
         i += (long)gridDim.x * 256) {
        float q = fminf(fmaxf(roundf(w[i] * ws), -1.f), 1.f) * inv;
        out[i] = (_Float16)q;
    }
}

// ---------------------------------------------------------------------------
// WMMA GEMM: C[M,N] = A[M,K](f16) x B[N,K](f16)^T, f32 accumulate.
// 256 threads = 8 wave32s; block tile 128(M) x 128(N); wave tile 32(M) x 64(N)
// (wave grid 4x2, 8 WMMA accumulators per wave).
// Requires M%128==0, N%128==0, K%32==0 (true for all shapes here).
// ---------------------------------------------------------------------------
__global__ __launch_bounds__(256) void gemm_f16_wmma_kernel(
    const _Float16* __restrict__ A, const _Float16* __restrict__ B,
    float* __restrict__ C, int M, int N, int K)
{
    __shared__ alignas(16) _Float16 As[128][32];  // 8 KB
    __shared__ alignas(16) _Float16 Bs[128][32];  // 8 KB

    const int tid  = threadIdx.x;
    const int lane = tid & 31;
    const int wave = tid >> 5;
    const int mlo  = lane & 15;   // row-in-tile (A) / col-in-tile (B,C)
    const int hi   = lane >> 4;   // lane-half select
    const int wm   = (wave >> 1) * 32;    // 0,32,64,96
    const int wn   = (wave & 1) * 64;     // 0,64
    const long bm  = (long)blockIdx.y * 128;
    const long bn  = (long)blockIdx.x * 128;

    // staging: 128 rows x 4 b128-chunks each for A and B; 2+2 chunks per thread
    const int row0 = tid >> 2;            // 0..63 (+64 for second chunk)
    const int col0 = (tid & 3) * 8;       // 0,8,16,24

    v8f acc[2][4];
#pragma unroll
    for (int r = 0; r < 2; ++r)
#pragma unroll
        for (int c = 0; c < 4; ++c)
#pragma unroll
            for (int v = 0; v < 8; ++v) acc[r][c][v] = 0.f;

    for (int k0 = 0; k0 < K; k0 += 32) {
        const _Float16* ga0 = A + (bm + row0) * (long)K + k0 + col0;
        const _Float16* ga1 = A + (bm + 64 + row0) * (long)K + k0 + col0;
        const _Float16* gb0 = B + (bn + row0) * (long)K + k0 + col0;
        const _Float16* gb1 = B + (bn + 64 + row0) * (long)K + k0 + col0;
#if HAVE_ASYNC_LDS
        // ASYNCcnt path: global -> LDS without bouncing through VGPRs
        __builtin_amdgcn_global_load_async_to_lds_b128(
            ASYNC_G(ga0), ASYNC_L(&As[row0][col0]), 0, 0);
        __builtin_amdgcn_global_load_async_to_lds_b128(
            ASYNC_G(ga1), ASYNC_L(&As[64 + row0][col0]), 0, 0);
        __builtin_amdgcn_global_load_async_to_lds_b128(
            ASYNC_G(gb0), ASYNC_L(&Bs[row0][col0]), 0, 0);
        __builtin_amdgcn_global_load_async_to_lds_b128(
            ASYNC_G(gb1), ASYNC_L(&Bs[64 + row0][col0]), 0, 0);
#else
        *(uint4*)(&As[row0][col0])      = *(const uint4*)ga0;
        *(uint4*)(&As[64 + row0][col0]) = *(const uint4*)ga1;
        *(uint4*)(&Bs[row0][col0])      = *(const uint4*)gb0;
        *(uint4*)(&Bs[64 + row0][col0]) = *(const uint4*)gb1;
#endif
        if (k0 + 32 < K) {
            __builtin_prefetch(ga0 + 32, 0, 1);
            __builtin_prefetch(ga1 + 32, 0, 1);
            __builtin_prefetch(gb0 + 32, 0, 1);
            __builtin_prefetch(gb1 + 32, 0, 1);
        }
#if HAVE_ASYNC_LDS
        wait_async_zero();
#endif
        __syncthreads();

        // A fragment: 16x32 f16, lane m = mlo; VGPR j holds
        // k = (j>>2)*16 + hi*8 + (j&3)*2 (+1) -> two 16B chunks at hi*8, 16+hi*8
        Frag af[2], bf[4];
#pragma unroll
        for (int r = 0; r < 2; ++r) {
            const _Float16* ap = &As[wm + r * 16 + mlo][0];
            af[r].q[0] = *(const uint4*)(ap + hi * 8);
            af[r].q[1] = *(const uint4*)(ap + 16 + hi * 8);
        }
        // B fragment: 32x16 f16, lane n = mlo; same k order as A:
        // two 16B chunks at hi*8, 16+hi*8
#pragma unroll
        for (int c = 0; c < 4; ++c) {
            const _Float16* bp = &Bs[wn + c * 16 + mlo][0];
            bf[c].q[0] = *(const uint4*)(bp + hi * 8);
            bf[c].q[1] = *(const uint4*)(bp + 16 + hi * 8);
        }
#pragma unroll
        for (int r = 0; r < 2; ++r)
#pragma unroll
            for (int c = 0; c < 4; ++c)
                acc[r][c] = __builtin_amdgcn_wmma_f32_16x16x32_f16(
                    false, af[r].h, false, bf[c].h,
                    (short)0, acc[r][c], false, false);
        __syncthreads();
    }

    // C/D layout: VGPR v, lane -> (row = v + hi*8, col = mlo) per 16x16 tile
#pragma unroll
    for (int r = 0; r < 2; ++r)
#pragma unroll
        for (int c = 0; c < 4; ++c)
#pragma unroll
            for (int v = 0; v < 8; ++v) {
                long row = bm + wm + r * 16 + hi * 8 + v;
                long col = bn + wn + c * 16 + mlo;
                C[row * (long)N + col] = acc[r][c][v];
            }
}

// ---------------------------------------------------------------------------
// Attention prep: split qkv rows into heads, per-head rms, head-indexed rotary
// (reference indexes the rotary table by the HEAD axis), head-major output.
// qh/kh/vh layout: [H][T][HD]
// ---------------------------------------------------------------------------
__global__ __launch_bounds__(64) void attnprep_kernel(
    const float* __restrict__ qkv, float* __restrict__ qh,
    float* __restrict__ kh, float* __restrict__ vh)
{
    int t = blockIdx.x, h = blockIdx.y, d = threadIdx.x;
    __shared__ float vec[3][HD_DIM];
    __shared__ float sh[HD_DIM];
    const float* row = qkv + (size_t)t * (3 * E_DIM) + h * HD_DIM;
    vec[0][d] = row[d];
    vec[1][d] = row[E_DIM + d];
    vec[2][d] = row[2 * E_DIM + d];
    for (int w = 0; w < 3; ++w) {
        __syncthreads();
        float v = vec[w][d];
        sh[d] = v * v;
        __syncthreads();
        for (int s2 = 32; s2 > 0; s2 >>= 1) {
            if (d < s2) sh[d] += sh[d + s2];
            __syncthreads();
        }
        float r = rsqrtf(sh[0] / (float)HD_DIM + RMS_EPS);
        vec[w][d] = v * r;
    }
    __syncthreads();
    size_t base = ((size_t)h * T_SEQ + t) * HD_DIM;
    if (d < HD_DIM / 2) {
        // freq[d] = 1024^(-d/15) for d<16, else 0 (theta indexed by head h)
        float fr = (d < HD_DIM / 4) ? powf(1.0f / 1024.0f, (float)d / 15.0f) : 0.0f;
        float th = (float)h * fr;
        float c = cosf(th), s = sinf(th);
        float x1 = vec[0][d], x2 = vec[0][d + 32];
        qh[base + d]      =  x1 * c + x2 * s;
        qh[base + d + 32] = -x1 * s + x2 * c;
        x1 = vec[1][d]; x2 = vec[1][d + 32];
        kh[base + d]      =  x1 * c + x2 * s;
        kh[base + d + 32] = -x1 * s + x2 * c;
    }
    vh[base + d] = vec[2][d];
}

// ---------------------------------------------------------------------------
// Causal attention core: one thread per (head, query), online softmax.
// y layout: [T][E] with head h occupying cols h*64..h*64+63
// ---------------------------------------------------------------------------
__global__ __launch_bounds__(256) void attn_core_kernel(
    const float* __restrict__ qh, const float* __restrict__ kh,
    const float* __restrict__ vh, float* __restrict__ y)
{
    int t = blockIdx.x * 256 + threadIdx.x;
    int h = blockIdx.y;
    const float* q = qh + ((size_t)h * T_SEQ + t) * HD_DIM;
    float qv[HD_DIM], acc[HD_DIM];
#pragma unroll
    for (int d = 0; d < HD_DIM; ++d) { qv[d] = q[d]; acc[d] = 0.f; }
    float m = -3.0e38f, ssum = 0.f;
    for (int j = 0; j <= t; ++j) {
        const float* kr = kh + ((size_t)h * T_SEQ + j) * HD_DIM;
        float s = 0.f;
#pragma unroll
        for (int d = 0; d < HD_DIM; ++d) s += qv[d] * kr[d];
        s *= ATTN_SCALE;
        float mn = fmaxf(m, s);
        float corr = expf(m - mn);
        float w = expf(s - mn);
        ssum = ssum * corr + w;
        const float* vr = vh + ((size_t)h * T_SEQ + j) * HD_DIM;
#pragma unroll
        for (int d = 0; d < HD_DIM; ++d) acc[d] = acc[d] * corr + w * vr[d];
        m = mn;
    }
    float inv = 1.0f / ssum;
    float* yo = y + (size_t)t * E_DIM + h * HD_DIM;
#pragma unroll
    for (int d = 0; d < HD_DIM; ++d) yo[d] = acc[d] * inv;
}

// ---------------------------------------------------------------------------
// Elementwise helpers
// ---------------------------------------------------------------------------
__global__ __launch_bounds__(256) void relusq_kernel(float* __restrict__ x, int n) {
    for (int i = blockIdx.x * blockDim.x + threadIdx.x; i < n;
         i += gridDim.x * blockDim.x) {
        float v = fmaxf(x[i], 0.f);
        x[i] = v * v;
    }
}

__global__ __launch_bounds__(256) void add3_kernel(
    float* __restrict__ x, const float* __restrict__ a,
    const float* __restrict__ b, const float* __restrict__ c, int n)
{
    int i = blockIdx.x * blockDim.x + threadIdx.x;
    if (i < n) x[i] += a[i] + b[i] + c[i];
}

// ---------------------------------------------------------------------------
// Loss: per-row logsumexp, then -mean(logp[target]) via atomicAdd
// ---------------------------------------------------------------------------
__global__ __launch_bounds__(256) void loss_kernel(
    const float* __restrict__ logits, const int* __restrict__ targets,
    float* __restrict__ loss)
{
    __shared__ float sh[256];
    int row = blockIdx.x;
    const float* p = logits + (size_t)row * V_VOCAB;
    float mx = -3.0e38f;
    for (int i = threadIdx.x; i < V_VOCAB; i += 256) mx = fmaxf(mx, p[i]);
    mx = block_reduce_max256(mx, sh);
    float se = 0.f;
    for (int i = threadIdx.x; i < V_VOCAB; i += 256) se += expf(p[i] - mx);
    se = block_reduce_sum256(se, sh);
    if (threadIdx.x == 0) {
        float lse = mx + logf(se);
        int tgt = targets[row];
        atomicAdd(loss, (lse - p[tgt]) * (1.0f / (float)T_SEQ));
    }
}

// ---------------------------------------------------------------------------
// Host-side orchestration
// ---------------------------------------------------------------------------
namespace {

struct Workspace {
    float*     x;       // [T,E]
    float*     xn;      // [T,E]
    float*     attn0;   // [T,E]
    float*     attn1;   // [T,E]
    float*     ffnout;  // [T,E]
    float*     ybuf;    // [T,E]
    float*     gbuf;    // [T,NH] (also qkv [T,3E])
    float*     qh;      // [H,T,HD]
    float*     kh;
    float*     vh;
    _Float16*  act16;   // [T,NH]
    _Float16*  wq16;    // max(V*E, NH*NH)
    float*     sumabs;  // 1 scalar
};

static void bitlinear(const float* x_in, int M, int K, const float* w, int N,
                      const Workspace& W, float* out, hipStream_t stream)
{
    // activation path: rms + int8 fake quant -> f16
    actquant_kernel<<<M, 256, 0, stream>>>(x_in, W.act16, K);
    // weight path: ternary fake quant -> f16
    zero_kernel<<<1, 32, 0, stream>>>(W.sumabs, 1);
    long n = (long)N * K;
    wabssum_kernel<<<1024, 256, 0, stream>>>(w, n, W.sumabs);
    wquant_kernel<<<1024, 256, 0, stream>>>(w, W.sumabs, n, W.wq16);
    // WMMA GEMM
    dim3 grid(N / 128, M / 128);
    gemm_f16_wmma_kernel<<<grid, 256, 0, stream>>>(W.act16, W.wq16, out, M, N, K);
}

} // namespace

extern "C" void kernel_launch(void* const* d_in, const int* in_sizes, int n_in,
                              void* d_out, int out_size, void* d_ws, size_t ws_size,
                              hipStream_t stream)
{
    const int*   idx     = (const int*)d_in[0];
    const int*   targets = (const int*)d_in[1];
    const float* wte     = (const float*)d_in[2];  // [V,E]
    const float* wpe     = (const float*)d_in[3];  // [T,E]
    const float* wqkv    = (const float*)d_in[4];  // [L,2,3E,E]
    const float* wproj   = (const float*)d_in[5];  // [L,2,E,E]
    const float* wfc1    = (const float*)d_in[6];  // [L,NH,E]
    const float* wfc2    = (const float*)d_in[7];  // [L,NH,NH]
    const float* wfcp    = (const float*)d_in[8];  // [L,E,NH]

    (void)in_sizes; (void)n_in; (void)out_size; (void)ws_size;

    // carve workspace
    char* p = (char*)d_ws;
    auto take = [&](size_t bytes) {
        void* q = p;
        p += (bytes + 255) & ~(size_t)255;
        return q;
    };
    Workspace W;
    const size_t TE = (size_t)T_SEQ * E_DIM;
    W.x      = (float*)take(TE * 4);
    W.xn     = (float*)take(TE * 4);
    W.attn0  = (float*)take(TE * 4);
    W.attn1  = (float*)take(TE * 4);
    W.ffnout = (float*)take(TE * 4);
    W.ybuf   = (float*)take(TE * 4);
    W.gbuf   = (float*)take((size_t)T_SEQ * NH_DIM * 4);
    W.qh     = (float*)take((size_t)H_HEADS * T_SEQ * HD_DIM * 4);
    W.kh     = (float*)take((size_t)H_HEADS * T_SEQ * HD_DIM * 4);
    W.vh     = (float*)take((size_t)H_HEADS * T_SEQ * HD_DIM * 4);
    W.act16  = (_Float16*)take((size_t)T_SEQ * NH_DIM * 2);
    W.wq16   = (_Float16*)take((size_t)V_VOCAB * E_DIM * 2);
    W.sumabs = (float*)take(256);

    float* logits = (float*)d_out;                       // [T, V]
    float* loss   = logits + (size_t)T_SEQ * V_VOCAB;    // scalar

    // ---- embedding ----
    embed_kernel<<<T_SEQ, 256, 0, stream>>>(idx, wte, wpe, W.x);

    // ---- transformer layers ----
    for (int l = 0; l < L_LAYERS; ++l) {
        rms_rows_kernel<<<T_SEQ, 256, 0, stream>>>(W.x, W.xn, E_DIM);

        // FFN: bitlinear -> relu^2 -> bitlinear -> relu^2 -> bitlinear
        bitlinear(W.xn, T_SEQ, E_DIM, wfc1 + (size_t)l * NH_DIM * E_DIM,
                  NH_DIM, W, W.gbuf, stream);
        relusq_kernel<<<4096, 256, 0, stream>>>(W.gbuf, T_SEQ * NH_DIM);
        bitlinear(W.gbuf, T_SEQ, NH_DIM, wfc2 + (size_t)l * NH_DIM * NH_DIM,
                  NH_DIM, W, W.gbuf, stream);
        relusq_kernel<<<4096, 256, 0, stream>>>(W.gbuf, T_SEQ * NH_DIM);
        bitlinear(W.gbuf, T_SEQ, NH_DIM, wfcp + (size_t)l * E_DIM * NH_DIM,
                  E_DIM, W, W.ffnout, stream);

        // two attention branches
        for (int a = 0; a < 2; ++a) {
            const float* wq = wqkv + ((size_t)(l * 2 + a)) * (3 * E_DIM) * E_DIM;
            const float* wp = wproj + ((size_t)(l * 2 + a)) * E_DIM * E_DIM;
            bitlinear(W.xn, T_SEQ, E_DIM, wq, 3 * E_DIM, W, W.gbuf, stream);
            attnprep_kernel<<<dim3(T_SEQ, H_HEADS), HD_DIM, 0, stream>>>(
                W.gbuf, W.qh, W.kh, W.vh);
            attn_core_kernel<<<dim3(T_SEQ / 256, H_HEADS), 256, 0, stream>>>(
                W.qh, W.kh, W.vh, W.ybuf);
            bitlinear(W.ybuf, T_SEQ, E_DIM, wp, E_DIM, W,
                      (a == 0) ? W.attn0 : W.attn1, stream);
        }

        add3_kernel<<<(T_SEQ * E_DIM) / 256, 256, 0, stream>>>(
            W.x, W.ffnout, W.attn0, W.attn1, T_SEQ * E_DIM);
    }

    // ---- lm head (tied to wte) + loss ----
    rms_rows_kernel<<<T_SEQ, 256, 0, stream>>>(W.x, W.xn, E_DIM);
    bitlinear(W.xn, T_SEQ, E_DIM, wte, V_VOCAB, W, logits, stream);
    zero_kernel<<<1, 32, 0, stream>>>(loss, 1);
    loss_kernel<<<T_SEQ, 256, 0, stream>>>(logits, targets, loss);
}
